// PointTransformerSeg_687194767477
// MI455X (gfx1250) — compile-verified, matrix operand order checked by reading, not run on hardware
//
#include <hip/hip_runtime.h>
#include <hip/hip_bf16.h>
#include <math.h>

typedef __attribute__((ext_vector_type(16))) _Float16 v16h;
typedef __attribute__((ext_vector_type(8)))  float    v8f;
typedef long long ll;

#define BN_EPS 1e-5f
#define LDS_PAD 8   // f16 elements; keeps rows 16B aligned, spreads LDS banks

// NOTE: __builtin_amdgcn_global_load_async_to_lds_b128 IS declared on this
// toolchain, but param0 is 'int __vector(4) *' (not an addrspace(1) void*).
// Full signature unknown from compile probes so far -> use the synchronous
// vector-copy staging path (global_load_b128 + ds_store_b128), which is safe.
#define HAVE_ASYNC_LDS 0

// ---------------------------------------------------------------------------
// GEMM: C[M,N] = act(A16[M,Kp] @ Wt[Np,Kp]^T + bias (+ residual))
// Block = 8 waves, block tile 128x64. The 64-column weight strip (64 x Kp f16,
// <= 64KB) is staged once into LDS, then each wave computes a 16x64 strip:
// 4 accumulators, A fragment reused 4x, B fragments read from LDS.
// ---------------------------------------------------------------------------
__global__ __launch_bounds__(256) void gemm_wmma_kernel(
    const _Float16* __restrict__ A, const _Float16* __restrict__ Bt,
    const float* __restrict__ bias, const float* __restrict__ residual,
    float* __restrict__ C, int M, int Kp, int N, int relu) {
  __shared__ __align__(16) _Float16 bs[64 * (512 + LDS_PAD)];
  const int tid  = threadIdx.x;
  const int lane = tid & 31;
  const int wave = tid >> 5;
  const int half = lane >> 4;
  const int l16  = lane & 15;
  const int strideL = Kp + LDS_PAD;
  const int nbase = blockIdx.y * 64;

  // ---- stage 64 x Kp weight strip into LDS ----
  {
    const int vecs   = (64 * Kp) >> 3;   // 16B chunks; Kp%32==0 -> vecs%256==0
    const int perRow = Kp >> 3;
    for (int i = tid; i < vecs; i += 256) {
      int row = i / perRow;
      int c8  = i % perRow;
      const _Float16* gsrc = Bt + (ll)(nbase + row) * Kp + (c8 << 3);
      _Float16* ldst = bs + row * strideL + (c8 << 3);
      *(uint4*)ldst = *(const uint4*)gsrc;
    }
  }
  __syncthreads();

  const int mtile = blockIdx.x * 128 + wave * 16;
  if (mtile >= M) return;              // wave-uniform; no barriers after this

  const _Float16* Ap = A + (ll)(mtile + l16) * Kp;
  v8f cacc[4];
#pragma unroll
  for (int j = 0; j < 4; ++j) cacc[j] = (v8f){0.f,0.f,0.f,0.f,0.f,0.f,0.f,0.f};

  for (int k0 = 0; k0 < Kp; k0 += 32) {
    __builtin_prefetch(Ap + k0 + 256, 0, 1);   // global_prefetch_b8, A stream
    v16h a;
#pragma unroll
    for (int i = 0; i < 8; ++i) {
      int k = k0 + ((i & 4) << 2) + (half << 3) + ((i & 3) << 1);
      a[2*i]   = Ap[k];
      a[2*i+1] = Ap[k+1];
    }
#pragma unroll
    for (int j = 0; j < 4; ++j) {
      const _Float16* Bp = bs + (j * 16 + l16) * strideL;
      v16h b;
#pragma unroll
      for (int i = 0; i < 8; ++i) {
        int k = k0 + ((i & 4) << 2) + (half << 3) + ((i & 3) << 1);
        b[2*i]   = Bp[k];
        b[2*i+1] = Bp[k+1];
      }
      cacc[j] = __builtin_amdgcn_wmma_f32_16x16x32_f16(false, a, false, b,
                                                       (short)0, cacc[j], false, false);
    }
  }

#pragma unroll
  for (int j = 0; j < 4; ++j) {
    int n = nbase + j * 16 + l16;
    if (n < N) {
      float bv = bias ? bias[n] : 0.f;
#pragma unroll
      for (int i = 0; i < 8; ++i) {
        int m = mtile + i + (half << 3);
        float v = cacc[j][i] + bv;
        if (residual) v += residual[(ll)m * N + n];
        if (relu) v = v > 0.f ? v : 0.f;
        C[(ll)m * N + n] = v;
      }
    }
  }
}

// Wt[n*Kp + k] = W[k*N + n], zero padded (Np padded to multiple of 64)
__global__ void prep_weight_kernel(const float* __restrict__ w, _Float16* __restrict__ wt,
                                   int K, int N, int Kp, int Np) {
  ll g = (ll)blockIdx.x * blockDim.x + threadIdx.x;
  ll total = (ll)Kp * Np;
  if (g >= total) return;
  int k = (int)(g % Kp);
  int n = (int)(g / Kp);
  wt[g] = (k < K && n < N) ? (_Float16)w[(ll)k * N + n] : (_Float16)0.f;
}

// dst[r*Kp + c] = (c<K) ? f16(src[r*K + c]) : 0
__global__ void cast_pad_kernel(const float* __restrict__ src, _Float16* __restrict__ dst,
                                int K, int Kp, ll total) {
  ll g = (ll)blockIdx.x * blockDim.x + threadIdx.x;
  if (g >= total) return;
  int c = (int)(g % Kp);
  ll r = g / Kp;
  dst[g] = (c < K) ? (_Float16)src[r * K + c] : (_Float16)0.f;
}

__global__ void slice_xyz_kernel(const float* __restrict__ x, float* __restrict__ xyz, ll total) {
  ll g = (ll)blockIdx.x * blockDim.x + threadIdx.x;
  if (g >= total) return;
  int c = (int)(g % 3);
  ll r = g / 3;
  xyz[g] = x[r * 6 + c];
}

// Farthest point sampling: one block per batch
__global__ void fps_kernel(const float* __restrict__ xyz, int* __restrict__ out,
                           int N, int npoint) {
  int b = blockIdx.x;
  int t = threadIdx.x;
  __shared__ float dist[2048];
  __shared__ float rv[256];
  __shared__ int   ri[256];
  __shared__ int   sfar;
  const float* p = xyz + (ll)b * N * 3;
  for (int n = t; n < N; n += 256) dist[n] = 1e10f;
  if (t == 0) sfar = 0;
  __syncthreads();
  for (int s = 0; s < npoint; ++s) {
    int far = sfar;
    if (t == 0) out[b * npoint + s] = far;
    float cx = p[far*3], cy = p[far*3+1], cz = p[far*3+2];
    float bestv = -1.f; int besti = 0;
    for (int n = t; n < N; n += 256) {
      float dx = p[n*3] - cx, dy = p[n*3+1] - cy, dz = p[n*3+2] - cz;
      float d = dx*dx + dy*dy + dz*dz;
      float dm = fminf(dist[n], d);
      dist[n] = dm;
      if (dm > bestv) { bestv = dm; besti = n; }
    }
    rv[t] = bestv; ri[t] = besti;
    __syncthreads();
    for (int off = 128; off > 0; off >>= 1) {
      if (t < off && rv[t + off] > rv[t]) { rv[t] = rv[t + off]; ri[t] = ri[t + off]; }
      __syncthreads();
    }
    if (t == 0) sfar = ri[0];
    __syncthreads();
  }
}

// brute-force k-NN (k <= 16), one thread per query
__global__ void knn_kernel(const float* __restrict__ q, const float* __restrict__ s,
                           int* __restrict__ idx, int Nq, int Ns, int k) {
  int gid = blockIdx.x * blockDim.x + threadIdx.x;
  int b = blockIdx.y;
  if (gid >= Nq) return;
  const float* qp = q + ((ll)b * Nq + gid) * 3;
  const float* sp = s + (ll)b * Ns * 3;
  float qx = qp[0], qy = qp[1], qz = qp[2];
  float bd[16]; int bi[16];
  for (int j = 0; j < 16; ++j) { bd[j] = 3.4e38f; bi[j] = 0; }
  for (int n = 0; n < Ns; ++n) {
    float dx = sp[n*3] - qx, dy = sp[n*3+1] - qy, dz = sp[n*3+2] - qz;
    float d = dx*dx + dy*dy + dz*dz;
    if (d < bd[k-1]) {
      int j = k - 1;
      while (j > 0 && bd[j-1] > d) { bd[j] = bd[j-1]; bi[j] = bi[j-1]; --j; }
      bd[j] = d; bi[j] = n;
    }
  }
  ll o = ((ll)b * Nq + gid) * k;
  for (int j = 0; j < k; ++j) idx[o + j] = bi[j];
}

// 3-NN with inverse-distance weights for interpolate()
__global__ void knn3_kernel(const float* __restrict__ q, const float* __restrict__ s,
                            int* __restrict__ idx, float* __restrict__ w, int Nq, int Ns) {
  int gid = blockIdx.x * blockDim.x + threadIdx.x;
  int b = blockIdx.y;
  if (gid >= Nq) return;
  const float* qp = q + ((ll)b * Nq + gid) * 3;
  const float* sp = s + (ll)b * Ns * 3;
  float bd[3] = {3.4e38f, 3.4e38f, 3.4e38f};
  int   bi[3] = {0, 0, 0};
  for (int n = 0; n < Ns; ++n) {
    float dx = sp[n*3] - qp[0], dy = sp[n*3+1] - qp[1], dz = sp[n*3+2] - qp[2];
    float d = dx*dx + dy*dy + dz*dz;
    if (d < bd[2]) {
      if (d < bd[0])      { bd[2]=bd[1]; bi[2]=bi[1]; bd[1]=bd[0]; bi[1]=bi[0]; bd[0]=d; bi[0]=n; }
      else if (d < bd[1]) { bd[2]=bd[1]; bi[2]=bi[1]; bd[1]=d; bi[1]=n; }
      else                { bd[2]=d; bi[2]=n; }
    }
  }
  float w0 = 1.f/(bd[0]+1e-8f), w1 = 1.f/(bd[1]+1e-8f), w2 = 1.f/(bd[2]+1e-8f);
  float wsum = w0 + w1 + w2;
  ll o = ((ll)b * Nq + gid) * 3;
  idx[o] = bi[0]; idx[o+1] = bi[1]; idx[o+2] = bi[2];
  w[o] = w0/wsum; w[o+1] = w1/wsum; w[o+2] = w2/wsum;
}

// dst[b,i,c] = src[b, idx[b,i], c]
__global__ void gather_kernel(const float* __restrict__ src, const int* __restrict__ idx,
                              float* __restrict__ dst, int Nq, int Ns, int C, ll total) {
  ll g = (ll)blockIdx.x * blockDim.x + threadIdx.x;
  if (g >= total) return;
  int c = (int)(g % C);
  ll r = g / C;                 // b*Nq + i
  int b = (int)(r / Nq);
  int j = idx[r];
  dst[g] = src[((ll)b * Ns + j) * C + c];
}

// delta input for fc_delta1: f16, K padded 3->32
__global__ void build_delta_kernel(const float* __restrict__ xyz, const int* __restrict__ idx,
                                   _Float16* __restrict__ out, int N, int k, ll rows) {
  ll r = (ll)blockIdx.x * blockDim.x + threadIdx.x;
  if (r >= rows) return;
  int b = (int)(r / ((ll)N * k));
  int n = (int)((r / k) % N);
  int j = idx[r];
  const float* pn = xyz + ((ll)b * N + n) * 3;
  const float* pj = xyz + ((ll)b * N + j) * 3;
  _Float16* o = out + r * 32;
  o[0] = (_Float16)(pn[0] - pj[0]);
  o[1] = (_Float16)(pn[1] - pj[1]);
  o[2] = (_Float16)(pn[2] - pj[2]);
#pragma unroll
  for (int c = 3; c < 32; ++c) o[c] = (_Float16)0.f;
}

// t16[r,c] = f16( q[b,n,c] - kf[b,idx,c] + pos[r,c] ), C = 512
__global__ void build_gamma_kernel(const float* __restrict__ q, const float* __restrict__ kf,
                                   const float* __restrict__ pos, const int* __restrict__ idx,
                                   _Float16* __restrict__ out, int N, int k, ll total) {
  ll g = (ll)blockIdx.x * blockDim.x + threadIdx.x;
  if (g >= total) return;
  int c = (int)(g & 511);
  ll r = g >> 9;
  int b = (int)(r / ((ll)N * k));
  int n = (int)((r / k) % N);
  int j = idx[r];
  float t = q[((ll)b * N + n) * 512 + c] - kf[((ll)b * N + j) * 512 + c] + pos[g];
  out[g] = (_Float16)t;
}

// softmax over the k neighbors per (b,n,c); C = 512
__global__ void softmax_kernel(float* __restrict__ a, int k, float scale, int /*N*/, ll total) {
  ll g = (ll)blockIdx.x * blockDim.x + threadIdx.x;
  if (g >= total) return;
  int c = (int)(g & 511);
  ll bn = g >> 9;
  ll base = bn * (ll)k * 512 + c;
  float vals[16];
  float mx = -3.4e38f;
  for (int i = 0; i < k; ++i) { float v = a[base + (ll)i * 512] * scale; vals[i] = v; mx = fmaxf(mx, v); }
  float s = 0.f;
  for (int i = 0; i < k; ++i) { float e = expf(vals[i] - mx); vals[i] = e; s += e; }
  float inv = 1.f / s;
  for (int i = 0; i < k; ++i) a[base + (ll)i * 512] = vals[i] * inv;
}

// res[b,n,c] = sum_k a[r,c] * (v[b,idx,c] + pos[r,c]); C = 512
__global__ void attn_out_kernel(const float* __restrict__ a, const float* __restrict__ v,
                                const float* __restrict__ pos, const int* __restrict__ idx,
                                float* __restrict__ res, int N, int k, ll total) {
  ll g = (ll)blockIdx.x * blockDim.x + threadIdx.x;
  if (g >= total) return;
  int c = (int)(g & 511);
  ll bn = g >> 9;
  int b = (int)(bn / N);
  float acc = 0.f;
  for (int i = 0; i < k; ++i) {
    ll r = bn * k + i;
    int j = idx[r];
    acc += a[r * 512 + c] * (v[((ll)b * N + j) * 512 + c] + pos[r * 512 + c]);
  }
  res[g] = acc;
}

// transition_down grouped feature: concat(xyz[idx]-new_xyz, points[idx]), f16 padded
__global__ void build_tdg_kernel(const float* __restrict__ xyz, const float* __restrict__ nxyz,
                                 const float* __restrict__ pts, const int* __restrict__ idx,
                                 _Float16* __restrict__ out, int np, int Ns, int Cin,
                                 int Kp, ll total) {
  ll g = (ll)blockIdx.x * blockDim.x + threadIdx.x;
  if (g >= total) return;
  int c = (int)(g % Kp);
  ll r = g / Kp;
  int b = (int)(r / ((ll)np * 16));
  int n = (int)((r / 16) % np);
  int j = idx[r];
  float v = 0.f;
  if (c < 3)            v = xyz[((ll)b * Ns + j) * 3 + c] - nxyz[((ll)b * np + n) * 3 + c];
  else if (c < 3 + Cin) v = pts[((ll)b * Ns + j) * Cin + (c - 3)];
  out[g] = (_Float16)v;
}

// per-channel mean/var (biased), one block per channel
__global__ void colstats_kernel(const float* __restrict__ x, float* __restrict__ mean,
                                float* __restrict__ var, ll rows, int C) {
  int c = blockIdx.x;
  int t = threadIdx.x;
  float s = 0.f, s2 = 0.f;
  for (ll r = t; r < rows; r += 256) { float v = x[r * C + c]; s += v; s2 += v * v; }
  __shared__ float ss[256];
  __shared__ float ss2[256];
  ss[t] = s; ss2[t] = s2;
  __syncthreads();
  for (int off = 128; off > 0; off >>= 1) {
    if (t < off) { ss[t] += ss[t + off]; ss2[t] += ss2[t + off]; }
    __syncthreads();
  }
  if (t == 0) {
    float m = ss[0] / (float)rows;
    mean[c] = m;
    var[c] = ss2[0] / (float)rows - m * m;
  }
}

__global__ void bn_relu_kernel(float* __restrict__ x, const float* __restrict__ mean,
                               const float* __restrict__ var, const float* __restrict__ gamma,
                               const float* __restrict__ beta, int C, ll total) {
  ll g = (ll)blockIdx.x * blockDim.x + threadIdx.x;
  if (g >= total) return;
  int c = (int)(g % C);
  float v = (x[g] - mean[c]) * rsqrtf(var[c] + BN_EPS) * gamma[c] + beta[c];
  x[g] = v > 0.f ? v : 0.f;
}

__global__ void maxk_kernel(const float* __restrict__ gdat, float* __restrict__ out,
                            int k, int C, ll total) {
  ll g = (ll)blockIdx.x * blockDim.x + threadIdx.x;
  if (g >= total) return;
  int c = (int)(g % C);
  ll row = g / C;
  float m = -3.4e38f;
  for (int i = 0; i < k; ++i) m = fmaxf(m, gdat[(row * k + i) * (ll)C + c]);
  out[g] = m;
}

__global__ void interp_kernel(const float* __restrict__ f1, const int* __restrict__ idx3,
                              const float* __restrict__ w3, const float* __restrict__ f2,
                              float* __restrict__ out, int N1, int N2, int C, ll total) {
  ll g = (ll)blockIdx.x * blockDim.x + threadIdx.x;
  if (g >= total) return;
  int c = (int)(g % C);
  ll r = g / C;                  // b*N2 + n
  int b = (int)(r / N2);
  ll o = r * 3;
  float acc = 0.f;
#pragma unroll
  for (int i = 0; i < 3; ++i)
    acc += w3[o + i] * f1[((ll)b * N1 + idx3[o + i]) * C + c];
  out[g] = acc + f2[g];
}

// ===========================================================================
extern "C" void kernel_launch(void* const* d_in, const int* in_sizes, int n_in,
                              void* d_out, int out_size, void* d_ws, size_t ws_size,
                              hipStream_t stream) {
  (void)in_sizes; (void)n_in; (void)out_size; (void)ws_size;

  const int numN[5] = {2048, 512, 128, 32, 8};
  const int chs[5]  = {32, 64, 128, 256, 512};

  // ---- parameter pytree walk (dict keys sorted alphabetically; x first) ----
  int pi = 1;
  auto nx = [&]() { return (const float*)d_in[pi++]; };
  struct TB { const float *fc1b,*fc1w,*fc2b,*fc2w,*d1b,*d1w,*d2b,*d2w,
                          *g1b,*g1w,*g2b,*g2w,*wks,*wqs,*wvs; };
  auto readTB = [&]() {
    TB t;
    t.fc1b=nx(); t.fc1w=nx(); t.fc2b=nx(); t.fc2w=nx();
    t.d1b=nx();  t.d1w=nx();  t.d2b=nx();  t.d2w=nx();
    t.g1b=nx();  t.g1w=nx();  t.g2b=nx();  t.g2w=nx();
    t.wks=nx();  t.wqs=nx();  t.wvs=nx();
    return t;
  };
  const float* xin = (const float*)d_in[0];
  TB dtfP[4]; for (int i = 0; i < 4; ++i) dtfP[i] = readTB();
  const float *fc1aB=nx(), *fc1aW=nx(), *fc1bB=nx(), *fc1bW=nx();
  const float *fc2aB=nx(), *fc2aW=nx(), *fc2bB=nx(), *fc2bW=nx(), *fc2cB=nx(), *fc2cW=nx();
  const float *fc3aB=nx(), *fc3aW=nx(), *fc3bB=nx(), *fc3bW=nx(), *fc3cB=nx(), *fc3cW=nx();
  struct TD { const float *b1,*beta1,*gamma1,*w1,*b2,*beta2,*gamma2,*w2; };
  TD tdP[4];
  for (int i = 0; i < 4; ++i) {
    tdP[i].b1=nx(); tdP[i].beta1=nx(); tdP[i].gamma1=nx(); tdP[i].w1=nx();
    tdP[i].b2=nx(); tdP[i].beta2=nx(); tdP[i].gamma2=nx(); tdP[i].w2=nx();
  }
  TB tfP[4]; for (int i = 0; i < 4; ++i) tfP[i] = readTB();
  TB tr1 = readTB();
  TB tr2 = readTB();
  struct TU { const float *beta1,*gamma1,*beta2,*gamma2,*fc1b,*fc1w,*fc2b,*fc2w; };
  TU tuP[4];
  for (int i = 0; i < 4; ++i) {
    tuP[i].beta1=nx(); tuP[i].gamma1=nx(); tuP[i].beta2=nx(); tuP[i].gamma2=nx();
    tuP[i].fc1b=nx();  tuP[i].fc1w=nx();  tuP[i].fc2b=nx();  tuP[i].fc2w=nx();
  }

  // ---- workspace bump allocator ----
  char* wsp = (char*)d_ws;
  size_t off = 0;
  auto wsa = [&](size_t bytes) -> void* {
    void* p = wsp + off;
    off += (bytes + 255) & ~(size_t)255;
    return p;
  };
  const ll MKMAX = 4LL * 2048 * 16;     // 131072
  const ll MMAX  = 4LL * 2048;          // 8192
  float*    BIGA = (float*)wsa((size_t)MKMAX * 512 * 4);
  float*    BIGB = (float*)wsa((size_t)MKMAX * 512 * 4);
  _Float16* BIGH = (_Float16*)wsa((size_t)MKMAX * 512 * 2);
  _Float16* H16  = (_Float16*)wsa((size_t)MMAX * 512 * 2);
  _Float16* DXH  = (_Float16*)wsa((size_t)MKMAX * 32 * 2);
  float* SM_X = (float*)wsa((size_t)MMAX * 512 * 4);
  float* SM_Q = (float*)wsa((size_t)MMAX * 512 * 4);
  float* SM_K = (float*)wsa((size_t)MMAX * 512 * 4);
  float* SM_V = (float*)wsa((size_t)MMAX * 512 * 4);
  float* GEN0 = (float*)wsa((size_t)MMAX * 512 * 4);
  float* GEN1 = (float*)wsa((size_t)MMAX * 512 * 4);
  _Float16* WT = (_Float16*)wsa((size_t)512 * 512 * 2);
  int*   IDXK = (int*)wsa((size_t)MKMAX * 4);
  int*   FPSI = (int*)wsa((size_t)4 * 512 * 4);
  int*   IDX3 = (int*)wsa((size_t)4 * 2048 * 3 * 4);
  float* W3   = (float*)wsa((size_t)4 * 2048 * 3 * 4);
  float* MEANB = (float*)wsa(512 * 4);
  float* VARB  = (float*)wsa(512 * 4);
  float* XYZL[5];
  float* PTSL[5];
  for (int i = 0; i < 5; ++i) {
    XYZL[i] = (float*)wsa((size_t)4 * numN[i] * 3 * 4);
    PTSL[i] = (float*)wsa((size_t)4 * numN[i] * chs[i] * 4);
  }
  float* TDTMP = (float*)wsa((size_t)4 * 512 * 64 * 4);
  float* MIDP  = (float*)wsa((size_t)4 * 8 * 512 * 4);
  float* DECP[4];
  for (int i = 0; i < 4; ++i) {
    int j = 3 - i;
    DECP[i] = (float*)wsa((size_t)4 * numN[j] * chs[j] * 4);
  }

  auto gb = [](ll t) { return (unsigned)((t + 255) / 256); };

  // ---- helpers ----
  auto cast16 = [&](const float* src, ll rows, int K, int Kp, _Float16* dst) {
    ll total = rows * (ll)Kp;
    cast_pad_kernel<<<gb(total), 256, 0, stream>>>(src, dst, K, Kp, total);
  };
  auto gemm16 = [&](const _Float16* A16, ll M, int K, const float* W, int N,
                    const float* bias, const float* resid, float* C, int relu) {
    int Kp = (K + 31) & ~31;
    int Np = (N + 63) & ~63;         // pad to the 64-wide LDS strip
    ll wt_total = (ll)Kp * Np;
    prep_weight_kernel<<<gb(wt_total), 256, 0, stream>>>(W, WT, K, N, Kp, Np);
    dim3 grid((unsigned)((M + 127) / 128), (unsigned)(Np / 64));
    gemm_wmma_kernel<<<grid, 256, 0, stream>>>(A16, WT, bias, resid, C, (int)M, Kp, N, relu);
  };

  const float attn_scale = 1.0f / sqrtf(512.0f);

  auto transformer = [&](const TB& p, const float* xyz, const float* feats,
                         float* outp, int N, int dp) {
    int kk = N < 16 ? N : 16;
    ll M = 4LL * N, Mk = M * kk;
    knn_kernel<<<dim3((unsigned)((N + 63) / 64), 4), 64, 0, stream>>>(xyz, xyz, IDXK, N, N, kk);
    // x = fc1(feats)
    cast16(feats, M, dp, dp, H16);
    gemm16(H16, M, dp, p.fc1w, 512, p.fc1b, nullptr, SM_X, 0);
    cast16(SM_X, M, 512, 512, H16);
    gemm16(H16, M, 512, p.wqs, 512, nullptr, nullptr, SM_Q, 0);
    gemm16(H16, M, 512, p.wks, 512, nullptr, nullptr, SM_K, 0);
    gemm16(H16, M, 512, p.wvs, 512, nullptr, nullptr, SM_V, 0);
    // pos = fc_delta2(relu(fc_delta1(dxyz)))
    build_delta_kernel<<<gb(Mk), 256, 0, stream>>>(xyz, IDXK, DXH, N, kk, Mk);
    gemm16(DXH, Mk, 3, p.d1w, 512, p.d1b, nullptr, BIGA, 1);
    cast16(BIGA, Mk, 512, 512, BIGH);
    gemm16(BIGH, Mk, 512, p.d2w, 512, p.d2b, nullptr, BIGB, 0);
    // a = fc_gamma2(relu(fc_gamma1(q - kf + pos)))
    build_gamma_kernel<<<gb(Mk * 512), 256, 0, stream>>>(SM_Q, SM_K, BIGB, IDXK, BIGH, N, kk, Mk * 512);
    gemm16(BIGH, Mk, 512, p.g1w, 512, p.g1b, nullptr, BIGA, 1);
    cast16(BIGA, Mk, 512, 512, BIGH);
    gemm16(BIGH, Mk, 512, p.g2w, 512, p.g2b, nullptr, BIGA, 0);
    softmax_kernel<<<gb(M * 512), 256, 0, stream>>>(BIGA, kk, attn_scale, N, M * 512);
    attn_out_kernel<<<gb(M * 512), 256, 0, stream>>>(BIGA, SM_V, BIGB, IDXK, SM_Q, N, kk, M * 512);
    // out = fc2(res) + feats
    cast16(SM_Q, M, 512, 512, H16);
    gemm16(H16, M, 512, p.fc2w, dp, p.fc2b, feats, outp, 0);
  };

  auto tdown = [&](const TD& p, const float* xyz, const float* pts, int Ns, int Cin,
                   int ch, float* nxyz, float* npts, int np) {
    fps_kernel<<<4, 256, 0, stream>>>(xyz, FPSI, Ns, np);
    gather_kernel<<<gb(4LL * np * 3), 256, 0, stream>>>(xyz, FPSI, nxyz, np, Ns, 3, 4LL * np * 3);
    knn_kernel<<<dim3((unsigned)((np + 63) / 64), 4), 64, 0, stream>>>(nxyz, xyz, IDXK, np, Ns, 16);
    ll rows = 4LL * np * 16;
    int K = Cin + 3, Kp = (K + 31) & ~31;
    build_tdg_kernel<<<gb(rows * Kp), 256, 0, stream>>>(xyz, nxyz, pts, IDXK, BIGH, np, Ns, Cin, Kp, rows * (ll)Kp);
    gemm16(BIGH, rows, K, p.w1, ch, p.b1, nullptr, BIGA, 0);
    colstats_kernel<<<ch, 256, 0, stream>>>(BIGA, MEANB, VARB, rows, ch);
    bn_relu_kernel<<<gb(rows * ch), 256, 0, stream>>>(BIGA, MEANB, VARB, p.gamma1, p.beta1, ch, rows * ch);
    cast16(BIGA, rows, ch, ch, BIGH);
    gemm16(BIGH, rows, ch, p.w2, ch, p.b2, nullptr, BIGA, 0);
    colstats_kernel<<<ch, 256, 0, stream>>>(BIGA, MEANB, VARB, rows, ch);
    bn_relu_kernel<<<gb(rows * ch), 256, 0, stream>>>(BIGA, MEANB, VARB, p.gamma2, p.beta2, ch, rows * ch);
    maxk_kernel<<<gb(4LL * np * ch), 256, 0, stream>>>(BIGA, npts, 16, ch, 4LL * np * ch);
  };

  auto tup = [&](const TU& p, const float* xyz1, const float* p1, int N1,
                 const float* xyz2, const float* p2, int N2, int ch, float* outp) {
    ll M1 = 4LL * N1, M2 = 4LL * N2;
    cast16(p1, M1, 2 * ch, 2 * ch, H16);
    gemm16(H16, M1, 2 * ch, p.fc1w, ch, p.fc1b, nullptr, SM_X, 0);
    colstats_kernel<<<ch, 256, 0, stream>>>(SM_X, MEANB, VARB, M1, ch);
    bn_relu_kernel<<<gb(M1 * ch), 256, 0, stream>>>(SM_X, MEANB, VARB, p.gamma1, p.beta1, ch, M1 * ch);
    cast16(p2, M2, ch, ch, H16);
    gemm16(H16, M2, ch, p.fc2w, ch, p.fc2b, nullptr, SM_K, 0);
    colstats_kernel<<<ch, 256, 0, stream>>>(SM_K, MEANB, VARB, M2, ch);
    bn_relu_kernel<<<gb(M2 * ch), 256, 0, stream>>>(SM_K, MEANB, VARB, p.gamma2, p.beta2, ch, M2 * ch);
    knn3_kernel<<<dim3((unsigned)((N2 + 63) / 64), 4), 64, 0, stream>>>(xyz2, xyz1, IDX3, W3, N2, N1);
    interp_kernel<<<gb(M2 * ch), 256, 0, stream>>>(SM_X, IDX3, W3, SM_K, outp, N1, N2, ch, M2 * ch);
  };

  // ================= forward =================
  // xyz = x[..., :3]
  slice_xyz_kernel<<<gb(4LL * 2048 * 3), 256, 0, stream>>>(xin, XYZL[0], 4LL * 2048 * 3);
  // h = fc1b(relu(fc1a(x)))
  cast16(xin, 8192, 6, 32, H16);
  gemm16(H16, 8192, 6, fc1aW, 32, fc1aB, nullptr, GEN0, 1);
  cast16(GEN0, 8192, 32, 32, H16);
  gemm16(H16, 8192, 32, fc1bW, 32, fc1bB, nullptr, GEN1, 0);
  // transformer1
  transformer(tr1, XYZL[0], GEN1, PTSL[0], 2048, 32);
  // encoder
  for (int i = 0; i < 4; ++i) {
    tdown(tdP[i], XYZL[i], PTSL[i], numN[i], chs[i], chs[i + 1], XYZL[i + 1], TDTMP, numN[i + 1]);
    transformer(tfP[i], XYZL[i + 1], TDTMP, PTSL[i + 1], numN[i + 1], chs[i + 1]);
  }
  // mid MLP + transformer2
  cast16(PTSL[4], 32, 512, 512, H16);
  gemm16(H16, 32, 512, fc2aW, 512, fc2aB, nullptr, GEN0, 1);
  cast16(GEN0, 32, 512, 512, H16);
  gemm16(H16, 32, 512, fc2bW, 512, fc2bB, nullptr, GEN1, 1);
  cast16(GEN1, 32, 512, 512, H16);
  gemm16(H16, 32, 512, fc2cW, 512, fc2cB, nullptr, GEN0, 0);
  transformer(tr2, XYZL[4], GEN0, MIDP, 8, 512);
  // decoder
  const float* cxyz = XYZL[4];
  const float* cpts = MIDP;
  int cN = 8;
  for (int i = 0; i < 4; ++i) {
    int j = 3 - i;
    int ch = chs[j];
    int N2 = numN[j];
    tup(tuP[i], cxyz, cpts, cN, XYZL[j], PTSL[j], N2, ch, DECP[i]);
    transformer(dtfP[i], XYZL[j], DECP[i], DECP[i], N2, ch);
    cxyz = XYZL[j];
    cpts = DECP[i];
    cN = N2;
  }
  // head: fc3c(relu(fc3b(relu(fc3a(points)))))
  cast16(DECP[3], 8192, 32, 32, H16);
  gemm16(H16, 8192, 32, fc3aW, 64, fc3aB, nullptr, GEN0, 1);
  cast16(GEN0, 8192, 64, 64, H16);
  gemm16(H16, 8192, 64, fc3bW, 64, fc3bB, nullptr, GEN1, 1);
  cast16(GEN1, 8192, 64, 64, H16);
  gemm16(H16, 8192, 64, fc3cW, 50, fc3cB, nullptr, (float*)d_out, 0);
}
